// NetDeconf_6511170421729
// MI455X (gfx1250) — matrix operand order repaired and checked by reading, not compiled, not run on hardware
//
#include <hip/hip_runtime.h>
#include <stdint.h>

typedef __attribute__((ext_vector_type(16))) _Float16 v16h;
typedef __attribute__((ext_vector_type(4)))  _Float16 v4h;
typedef __attribute__((ext_vector_type(2)))  _Float16 v2h;
typedef __attribute__((ext_vector_type(8)))  float    v8f;

#define N_NODES    50000
#define N_EDGES    800000
#define NFEAT      128
#define NHID       64
#define N_PATIENTS 10000

// ---------------------------------------------------------------------------
// V_WMMA_F32_16X16X32_F16 fragment layouts as used here (contiguous k, not the ISA table order):
//  A (16x32 f16): lane = half*16 + m holds row m; slot e -> K = half*16 + e
//    (16 contiguous, matching B) -> 4x b128 loads.
//  B (32x16 f16, same k order as A): lane = half*16 + n holds column n,
//    slot e -> K = half*16 + e (16 contiguous) -> one 32B load from f16
//    transposed weights Wt[n][k].
//  C/D (16x16 f32): VGPR r -> row half*8+r, lane%16 -> column.
// ---------------------------------------------------------------------------

template <bool USE_BIAS>
__device__ __forceinline__ v16h load_a_frag(const float* __restrict__ inRow,
                                            const float* __restrict__ bias,
                                            int k0, int half) {
    const int kb = k0 + half * 16;
    float4 r0 = *(const float4*)(inRow + kb);
    float4 r1 = *(const float4*)(inRow + kb + 4);
    float4 r2 = *(const float4*)(inRow + kb + 8);
    float4 r3 = *(const float4*)(inRow + kb + 12);
    float v[16] = {r0.x, r0.y, r0.z, r0.w, r1.x, r1.y, r1.z, r1.w,
                   r2.x, r2.y, r2.z, r2.w, r3.x, r3.y, r3.z, r3.w};
    if (USE_BIAS) {
        float4 b0 = *(const float4*)(bias + kb);
        float4 b1 = *(const float4*)(bias + kb + 4);
        float4 b2 = *(const float4*)(bias + kb + 8);
        float4 b3 = *(const float4*)(bias + kb + 12);
        float bv[16] = {b0.x, b0.y, b0.z, b0.w, b1.x, b1.y, b1.z, b1.w,
                        b2.x, b2.y, b2.z, b2.w, b3.x, b3.y, b3.z, b3.w};
#pragma unroll
        for (int i = 0; i < 16; ++i) v[i] = fmaxf(v[i] + bv[i], 0.0f);
    }
    v16h a;
#pragma unroll
    for (int e = 0; e < 16; ++e) a[e] = (_Float16)v[e];
    return a;
}

// out[M x 64] (f16) = act(in)[M x K] @ W[K x 64]; Wt is f16 [64][K] (transposed).
// act(x) = USE_BIAS ? relu(x + bias[k]) : x. One wave per 16-row tile.
template <int K, bool USE_BIAS>
__global__ void gemm_wmma_kernel(const float* __restrict__ in,
                                 const float* __restrict__ bias,
                                 const _Float16* __restrict__ Wt,
                                 _Float16* __restrict__ out) {
    const int wave = blockIdx.x * 5 + (threadIdx.x >> 5);  // 625*5 = 3125 tiles exact
    const int lane = threadIdx.x & 31;
    const int m = lane & 15, half = lane >> 4;
    const int rowBase = wave << 4;
    const float* inRow = in + (size_t)(rowBase + m) * K;

    v8f acc[4] = {};
#pragma unroll
    for (int k0 = 0; k0 < K; k0 += 32) {
        v16h a = load_a_frag<USE_BIAS>(inRow, bias, k0, half);
#pragma unroll
        for (int ct = 0; ct < 4; ++ct) {
            const int n = ct * 16 + m;
            v16h b = *(const v16h*)(Wt + (size_t)n * K + k0 + half * 16);
            acc[ct] = __builtin_amdgcn_wmma_f32_16x16x32_f16(
                false, a, false, b, (short)0, acc[ct], false, false);
        }
    }
#pragma unroll
    for (int ct = 0; ct < 4; ++ct)
#pragma unroll
        for (int r = 0; r < 8; ++r)
            out[(size_t)(rowBase + half * 8 + r) * NHID + ct * 16 + m] =
                (_Float16)acc[ct][r];
}

// ---------------------------------------------------------------------------
// Pack weights: W [K][64] f32 -> Wt [64][K] f16 (transposed), done once.
__global__ void pack_w_kernel(const float* __restrict__ W, _Float16* __restrict__ Wt,
                              int K) {
    int idx = blockIdx.x * 256 + threadIdx.x;
    if (idx < K * NHID) {
        int n = idx / K, k = idx % K;
        Wt[idx] = (_Float16)W[(size_t)k * NHID + n];
    }
}

// ---------------------------------------------------------------------------
// Degree / normalization
__global__ void deg_init_kernel(float* __restrict__ deg) {
    int i = blockIdx.x * 256 + threadIdx.x;
    if (i < N_NODES) deg[i] = 1.0f;  // self-loop
}
__global__ void deg_accum_kernel(const int* __restrict__ dst, float* __restrict__ deg) {
    int e = blockIdx.x * 256 + threadIdx.x;
    if (e < N_EDGES) atomicAdd(&deg[dst[e]], 1.0f);
}
__global__ void dinv_kernel(const float* __restrict__ deg, float* __restrict__ dinv) {
    int i = blockIdx.x * 256 + threadIdx.x;
    if (i < N_NODES) dinv[i] = rsqrtf(deg[i]);
}

// ---------------------------------------------------------------------------
// out(f32) = segsum_{j->i} dinv_j*dinv_i*h_j(f16)  (+ self loop dinv_i^2*h_i)
__global__ void agg_init_kernel(const _Float16* __restrict__ h,
                                const float* __restrict__ dinv,
                                float* __restrict__ out) {
    unsigned idx = blockIdx.x * 256u + threadIdx.x;  // over N*16 float4 groups
    if (idx < (unsigned)N_NODES * (NHID / 4)) {
        unsigned node = idx >> 4;
        float d = dinv[node];
        float s = d * d;
        v4h hv = *(const v4h*)(h + (size_t)idx * 4);
        float4 o = {(float)hv[0] * s, (float)hv[1] * s,
                    (float)hv[2] * s, (float)hv[3] * s};
        ((float4*)out)[idx] = o;  // also re-initializes poisoned scratch
    }
}
__global__ void agg_edge_kernel(const _Float16* __restrict__ h,
                                const int* __restrict__ src,
                                const int* __restrict__ dst,
                                const float* __restrict__ dinv,
                                float* __restrict__ out) {
    unsigned idx = blockIdx.x * 256u + threadIdx.x;  // over E*32 half2 groups
    if (idx < (unsigned)N_EDGES * (NHID / 2)) {
        unsigned e = idx >> 5, f2 = idx & 31;
        int s = src[e], d = dst[e];
        float w = dinv[s] * dinv[d];
        v2h hv = *(const v2h*)(h + (size_t)s * NHID + f2 * 2);
        float* o = out + (size_t)d * NHID + f2 * 2;
        atomicAdd(o, (float)hv[0] * w);
        atomicAdd(o + 1, (float)hv[1] * w);
    }
}

// ---------------------------------------------------------------------------
// Fused heads: din = relu(in + gc1_b); per branch:
//   y = sigmoid( relu(din @ W0 + b0) @ w1 + b1 )
__global__ void head_wmma_kernel(const float* __restrict__ in,
                                 const float* __restrict__ in_bias,
                                 const _Float16* __restrict__ Wt0a,
                                 const float* __restrict__ b0a,
                                 const _Float16* __restrict__ Wt0b,
                                 const float* __restrict__ b0b,
                                 const float* __restrict__ w1a,
                                 const float* __restrict__ b1a,
                                 const float* __restrict__ w1b,
                                 const float* __restrict__ b1b,
                                 float* __restrict__ y0, float* __restrict__ y1) {
    const int wave = blockIdx.x * 5 + (threadIdx.x >> 5);
    const int lane = threadIdx.x & 31;
    const int m = lane & 15, half = lane >> 4;
    const int rowBase = wave << 4;
    const float* inRow = in + (size_t)(rowBase + m) * NHID;

    v8f acc0[4] = {}, acc1[4] = {};
#pragma unroll
    for (int k0 = 0; k0 < NHID; k0 += 32) {
        v16h a = load_a_frag<true>(inRow, in_bias, k0, half);
#pragma unroll
        for (int ct = 0; ct < 4; ++ct) {
            const int n = ct * 16 + m;
            size_t off = (size_t)n * NHID + k0 + half * 16;
            v16h ba = *(const v16h*)(Wt0a + off);
            v16h bb = *(const v16h*)(Wt0b + off);
            acc0[ct] = __builtin_amdgcn_wmma_f32_16x16x32_f16(
                false, a, false, ba, (short)0, acc0[ct], false, false);
            acc1[ct] = __builtin_amdgcn_wmma_f32_16x16x32_f16(
                false, a, false, bb, (short)0, acc1[ct], false, false);
        }
    }
    // partial dot with 64x1 projection, reduce across 16 lanes holding a row
    float p0[8] = {}, p1[8] = {};
#pragma unroll
    for (int ct = 0; ct < 4; ++ct) {
        const int col = ct * 16 + m;
        const float wa = w1a[col], wb = w1b[col];
        const float c0 = b0a[col], c1 = b0b[col];
#pragma unroll
        for (int r = 0; r < 8; ++r) {
            p0[r] += fmaxf(acc0[ct][r] + c0, 0.0f) * wa;
            p1[r] += fmaxf(acc1[ct][r] + c1, 0.0f) * wb;
        }
    }
#pragma unroll
    for (int off = 8; off >= 1; off >>= 1)
#pragma unroll
        for (int r = 0; r < 8; ++r) {
            p0[r] += __shfl_xor(p0[r], off, 16);
            p1[r] += __shfl_xor(p1[r], off, 16);
        }
    if (m == 0) {
        const float bb1a = b1a[0], bb1b = b1b[0];
#pragma unroll
        for (int r = 0; r < 8; ++r) {
            int row = rowBase + half * 8 + r;
            y0[row] = 1.0f / (1.0f + expf(-(p0[r] + bb1a)));
            y1[row] = 1.0f / (1.0f + expf(-(p1[r] + bb1b)));
        }
    }
}

// ---------------------------------------------------------------------------
__global__ void gather_kernel(const float* __restrict__ y0, const float* __restrict__ y1,
                              const int* __restrict__ pid, const int* __restrict__ treat,
                              float* __restrict__ out) {
    int p = blockIdx.x * 256 + threadIdx.x;
    if (p < N_PATIENTS) {
        int n = pid[p];
        float a = y0[n], b = y1[n];
        out[p] = (treat[p] > 0) ? b : a;  // y
        out[N_PATIENTS + p] = b;          // y1p
        out[2 * N_PATIENTS + p] = a;      // y0p
    }
}

// ---------------------------------------------------------------------------
extern "C" void kernel_launch(void* const* d_in, const int* in_sizes, int n_in,
                              void* d_out, int out_size, void* d_ws, size_t ws_size,
                              hipStream_t stream) {
    const float* x     = (const float*)d_in[0];
    const int*   ei    = (const int*)d_in[1];  // [2, E] flat
    const int*   pid   = (const int*)d_in[2];
    const int*   treat = (const int*)d_in[3];
    const float* gc0_W = (const float*)d_in[4];
    const float* gc0_b = (const float*)d_in[5];
    const float* gc1_W = (const float*)d_in[6];
    const float* gc1_b = (const float*)d_in[7];
    const float* t00_W = (const float*)d_in[8];
    const float* t00_b = (const float*)d_in[9];
    const float* t10_W = (const float*)d_in[10];
    const float* t10_b = (const float*)d_in[11];
    const float* t01_W = (const float*)d_in[12];
    const float* t01_b = (const float*)d_in[13];
    const float* t11_W = (const float*)d_in[14];
    const float* t11_b = (const float*)d_in[15];
    const int* src = ei;
    const int* dst = ei + N_EDGES;

    float* deg  = (float*)d_ws;                    // N
    float* dinv = deg + N_NODES;                   // N
    float* Ab   = dinv + N_NODES;                  // N*64 f32 (layer-0 aggregated)
    float* Bb   = Ab + (size_t)N_NODES * NHID;     // N*64 f32 (layer-1 aggregated)
    float* y0b  = Bb + (size_t)N_NODES * NHID;     // N
    float* y1b  = y0b + N_NODES;                   // N
    _Float16* Hh   = (_Float16*)(y1b + N_NODES);   // N*64 f16 (pre-aggregation XW)
    _Float16* Wt0  = Hh + (size_t)N_NODES * NHID;  // 64*128 f16
    _Float16* Wt1  = Wt0 + NHID * NFEAT;           // 64*64 f16
    _Float16* Wt00 = Wt1 + NHID * NHID;
    _Float16* Wt10 = Wt00 + NHID * NHID;

    const int gN   = (N_NODES + 255) / 256;
    const int gE   = (N_EDGES + 255) / 256;
    const unsigned gNI = ((unsigned)N_NODES * (NHID / 4) + 255u) / 256u;
    const unsigned gEA = ((unsigned)N_EDGES * (NHID / 2) + 255u) / 256u;
    const int gP   = (N_PATIENTS + 255) / 256;
    const dim3 gemmGrid(625), gemmBlk(160);  // 3125 waves == 3125 row tiles (exact)

    // one-shot weight packing (f32 [K][64] -> f16 [64][K])
    pack_w_kernel<<<(NFEAT * NHID + 255) / 256, 256, 0, stream>>>(gc0_W, Wt0, NFEAT);
    pack_w_kernel<<<(NHID * NHID + 255) / 256, 256, 0, stream>>>(gc1_W, Wt1, NHID);
    pack_w_kernel<<<(NHID * NHID + 255) / 256, 256, 0, stream>>>(t00_W, Wt00, NHID);
    pack_w_kernel<<<(NHID * NHID + 255) / 256, 256, 0, stream>>>(t10_W, Wt10, NHID);

    // normalization
    deg_init_kernel<<<gN, 256, 0, stream>>>(deg);
    deg_accum_kernel<<<gE, 256, 0, stream>>>(dst, deg);
    dinv_kernel<<<gN, 256, 0, stream>>>(deg, dinv);

    // layer 0: H = X @ W0 ; A = aggregate(H)
    gemm_wmma_kernel<NFEAT, false><<<gemmGrid, gemmBlk, 0, stream>>>(x, nullptr, Wt0, Hh);
    agg_init_kernel<<<gNI, 256, 0, stream>>>(Hh, dinv, Ab);
    agg_edge_kernel<<<gEA, 256, 0, stream>>>(Hh, src, dst, dinv, Ab);

    // layer 1: H = relu(A + b0) @ W1 ; B = aggregate(H)
    gemm_wmma_kernel<NHID, true><<<gemmGrid, gemmBlk, 0, stream>>>(Ab, gc0_b, Wt1, Hh);
    agg_init_kernel<<<gNI, 256, 0, stream>>>(Hh, dinv, Bb);
    agg_edge_kernel<<<gEA, 256, 0, stream>>>(Hh, src, dst, dinv, Bb);

    // fused heads + sigmoid
    head_wmma_kernel<<<gemmGrid, gemmBlk, 0, stream>>>(Bb, gc1_b, Wt00, t00_b,
                                                       Wt10, t10_b, t01_W, t01_b,
                                                       t11_W, t11_b, y0b, y1b);

    // patient gather / treatment select
    gather_kernel<<<gP, 256, 0, stream>>>(y0b, y1b, pid, treat, (float*)d_out);
}
